// DeepJ_335007449482
// MI455X (gfx1250) — compile-verified, matrix operand order checked by reading, not run on hardware
//
#include <hip/hip_runtime.h>
#include <math.h>

#define B_    1024
#define N_    48
#define TU_   256
#define NU_   128
#define ROWS  (B_ * N_)     // 49152
#define K1    64            // IN_T=50 padded to 64

typedef __attribute__((ext_vector_type(16))) _Float16 v16h;
typedef __attribute__((ext_vector_type(8)))  float    v8f;

union FragU { v16h h; uint4 u4[2]; };

// Load a 16x32 f16 A/B fragment for one lane from a row-major row pointer
// (row stride handled by caller). Per CDNA5 ISA 16-bit A layout:
// lanes 0-15 hold K = k0+{0..7} and k0+{16..23}; lanes 16-31 hold +8.
__device__ __forceinline__ v16h ldfrag(const _Float16* rowp, int k0, int lane) {
    FragU f;
    int kb = k0 + ((lane & 16) ? 8 : 0);
    f.u4[0] = *(const uint4*)(rowp + kb);
    f.u4[1] = *(const uint4*)(rowp + kb + 16);
    return f.h;
}

__device__ __forceinline__ v8f wmma_f16(v16h a, v16h b, v8f c) {
    return __builtin_amdgcn_wmma_f32_16x16x32_f16(false, a, false, b,
                                                  (short)0, c, false, false);
}

__device__ __forceinline__ float sigm(float x) { return 1.0f / (1.0f + __expf(-x)); }

// ---------------------------------------------------------------------------
// Build rnn_in f16 [ROWS x 64]: [pitch_pos, pitch_class(12), vicinity(25),
// chord(12), pad(14)]
// ---------------------------------------------------------------------------
__global__ void build_rnn_in_kernel(const float* __restrict__ note,
                                    _Float16* __restrict__ out) {
    int idx = blockIdx.x * blockDim.x + threadIdx.x;
    if (idx >= ROWS * K1) return;
    int row = idx >> 6;
    int k   = idx & 63;
    int b   = row / N_;
    int n   = row % N_;
    float v = 0.0f;
    if (k == 0) {
        v = (float)n * (1.0f / (float)N_);
    } else if (k < 13) {
        v = ((k - 1) == (n % 12)) ? 1.0f : 0.0f;
    } else if (k < 38) {
        int off = n + (k - 13) - 12;            // vicinity
        v = (off >= 0 && off < N_) ? note[b * N_ + off] : 0.0f;
    } else if (k < 50) {
        int o = k - 38;                         // chord: sum over 4 in reshape(12,4)
        const float* nb = note + b * N_ + o * 4;
        v = nb[0] + nb[1] + nb[2] + nb[3];
    }
    out[idx] = (_Float16)v;
}

// ---------------------------------------------------------------------------
// Generic f32 -> f16 weight convert with K-padding / K-truncation
// ---------------------------------------------------------------------------
__global__ void cvt_pad_kernel(const float* __restrict__ src, _Float16* __restrict__ dst,
                               int rows, int sstride, int ccols, int dcols) {
    int idx = blockIdx.x * blockDim.x + threadIdx.x;
    if (idx >= rows * dcols) return;
    int r = idx / dcols, c = idx % dcols;
    float v = (c < ccols) ? src[(size_t)r * sstride + c] : 0.0f;
    dst[idx] = (_Float16)v;
}

__global__ void extract_col_kernel(const float* __restrict__ src, float* __restrict__ dst,
                                   int rows, int stride, int colidx) {
    int i = blockIdx.x * blockDim.x + threadIdx.x;
    if (i < rows) dst[i] = src[(size_t)i * stride + colidx];
}

__global__ void zero_state_kernel(_Float16* h1, _Float16* h2, float* c1, float* c2) {
    int i = blockIdx.x * blockDim.x + threadIdx.x;
    if (i < B_ * NU_) {
        h1[i] = (_Float16)0.0f; h2[i] = (_Float16)0.0f;
        c1[i] = 0.0f;           c2[i] = 0.0f;
    }
}

// ---------------------------------------------------------------------------
// Time-axis LSTM layer with zero initial state: H = sig(o)*tanh(sig(i)*tanh(g))
// A [ROWS x K] f16, W [1024 x K] f16 (row n = gate-output n), H [ROWS x 256].
// f-gate is dead (c_prev==0). One wave per 16x16 output tile, 3 WMMA chains.
// ---------------------------------------------------------------------------
__global__ void time_lstm_kernel(const _Float16* __restrict__ A,
                                 const _Float16* __restrict__ W,
                                 const float* __restrict__ bias,
                                 _Float16* __restrict__ H, int K) {
    int gw   = (blockIdx.x * blockDim.x + threadIdx.x) >> 5;
    int lane = threadIdx.x & 31;
    int mt = gw >> 4;           // 16 unit-tiles per row-tile
    int ut = gw & 15;
    int m0 = mt * 16, u0 = ut * 16;
    int col = lane & 15;
    int hi  = (lane & 16) ? 8 : 0;
    if (m0 >= ROWS) return;

    const _Float16* arow = A + (size_t)(m0 + col) * K;
    const _Float16* wir  = W + (size_t)(0 * TU_ + u0 + col) * K;  // i gate
    const _Float16* wgr  = W + (size_t)(2 * TU_ + u0 + col) * K;  // g gate
    const _Float16* wor  = W + (size_t)(3 * TU_ + u0 + col) * K;  // o gate

    v8f ai = {}, ag = {}, ao = {};
    for (int k0 = 0; k0 < K; k0 += 32) {
        v16h a = ldfrag(arow, k0, lane);
        ai = wmma_f16(a, ldfrag(wir, k0, lane), ai);
        ag = wmma_f16(a, ldfrag(wgr, k0, lane), ag);
        ao = wmma_f16(a, ldfrag(wor, k0, lane), ao);
    }
    int u = u0 + col;
    float bi = bias[u], bg = bias[2 * TU_ + u], bo = bias[3 * TU_ + u];
#pragma unroll
    for (int r = 0; r < 8; ++r) {
        int m = m0 + r + hi;
        float c = sigm(ai[r] + bi) * tanhf(ag[r] + bg);
        float h = sigm(ao[r] + bo) * tanhf(c);
        H[(size_t)m * TU_ + u] = (_Float16)h;
    }
}

// ---------------------------------------------------------------------------
// One note-axis scan step: both LSTM cells + output projection fused.
// Block = 16 batch rows x 8 waves (each wave owns one 16-unit tile, 4 gates).
// ---------------------------------------------------------------------------
__global__ void note_step_kernel(
    const _Float16* __restrict__ feats,   // [B, N, 256] f16
    const float*    __restrict__ targets, // [B, N]
    const _Float16* __restrict__ Wih0,    // [512, 256]
    const float*    __restrict__ wlast,   // [512]   (Wih0 column 256, f32)
    const _Float16* __restrict__ Whh0,    // [512, 128]
    const float*    __restrict__ b0,      // [512]
    const _Float16* __restrict__ Wih1,    // [512, 128]
    const _Float16* __restrict__ Whh1,    // [512, 128]
    const float*    __restrict__ b1,      // [512]
    const float*    __restrict__ outW,    // [128]
    const float*    __restrict__ outb,    // [1]
    _Float16* __restrict__ h1s, float* __restrict__ c1s,
    _Float16* __restrict__ h2s, float* __restrict__ c2s,
    float* __restrict__ out, int step) {
    __shared__ __align__(16) _Float16 h1_l[16 * NU_];
    __shared__ __align__(16) _Float16 h2_l[16 * NU_];
    __shared__ float cond_s[16];

    const int tid  = threadIdx.x;
    const int lane = tid & 31;
    const int w    = tid >> 5;           // wave 0..7
    const int m0   = blockIdx.x * 16;
    const int u0   = w * 16;
    const int col  = lane & 15;
    const int hi   = (lane & 16) ? 8 : 0;
    const v8f vzero = {};

    if (tid < 16) {
        int m = m0 + tid;
        cond_s[tid] = (step == 0) ? 0.0f : targets[(size_t)m * N_ + step - 1];
    }
    __syncthreads();

    // ---------------- cell 0 ----------------
    v8f a0 = vzero, a1 = vzero, a2 = vzero, a3 = vzero;
    {   // x (feats) @ Wih0[:, 0:256].T  -- K = 256
        const _Float16* arow = feats + ((size_t)(m0 + col) * N_ + step) * TU_;
        const _Float16* w0r  = Wih0 + (size_t)(0 * NU_ + u0 + col) * TU_;
        const _Float16* w1r  = Wih0 + (size_t)(1 * NU_ + u0 + col) * TU_;
        const _Float16* w2r  = Wih0 + (size_t)(2 * NU_ + u0 + col) * TU_;
        const _Float16* w3r  = Wih0 + (size_t)(3 * NU_ + u0 + col) * TU_;
        for (int k0 = 0; k0 < TU_; k0 += 32) {
            v16h a = ldfrag(arow, k0, lane);
            a0 = wmma_f16(a, ldfrag(w0r, k0, lane), a0);
            a1 = wmma_f16(a, ldfrag(w1r, k0, lane), a1);
            a2 = wmma_f16(a, ldfrag(w2r, k0, lane), a2);
            a3 = wmma_f16(a, ldfrag(w3r, k0, lane), a3);
        }
    }
    {   // h1 @ Whh0.T  -- K = 128
        const _Float16* arow = h1s + (size_t)(m0 + col) * NU_;
        const _Float16* w0r  = Whh0 + (size_t)(0 * NU_ + u0 + col) * NU_;
        const _Float16* w1r  = Whh0 + (size_t)(1 * NU_ + u0 + col) * NU_;
        const _Float16* w2r  = Whh0 + (size_t)(2 * NU_ + u0 + col) * NU_;
        const _Float16* w3r  = Whh0 + (size_t)(3 * NU_ + u0 + col) * NU_;
        for (int k0 = 0; k0 < NU_; k0 += 32) {
            v16h a = ldfrag(arow, k0, lane);
            a0 = wmma_f16(a, ldfrag(w0r, k0, lane), a0);
            a1 = wmma_f16(a, ldfrag(w1r, k0, lane), a1);
            a2 = wmma_f16(a, ldfrag(w2r, k0, lane), a2);
            a3 = wmma_f16(a, ldfrag(w3r, k0, lane), a3);
        }
    }
    {   // rank-1 cond term (257th input column)
        float wl0 = wlast[0 * NU_ + u0 + col];
        float wl1 = wlast[1 * NU_ + u0 + col];
        float wl2 = wlast[2 * NU_ + u0 + col];
        float wl3 = wlast[3 * NU_ + u0 + col];
#pragma unroll
        for (int r = 0; r < 8; ++r) {
            float cd = cond_s[r + hi];
            a0[r] += cd * wl0; a1[r] += cd * wl1;
            a2[r] += cd * wl2; a3[r] += cd * wl3;
        }
    }
    __syncthreads();   // all waves done reading h1s before we overwrite it
    {
        int u = u0 + col;
        float bi = b0[u], bf = b0[NU_ + u], bg = b0[2 * NU_ + u], bo = b0[3 * NU_ + u];
#pragma unroll
        for (int r = 0; r < 8; ++r) {
            int mr = r + hi;
            int m  = m0 + mr;
            float cprev = c1s[(size_t)m * NU_ + u];
            float c = sigm(a1[r] + bf) * cprev + sigm(a0[r] + bi) * tanhf(a2[r] + bg);
            float h = sigm(a3[r] + bo) * tanhf(c);
            c1s[(size_t)m * NU_ + u] = c;
            _Float16 hh = (_Float16)h;
            h1_l[mr * NU_ + u]       = hh;   // for cell 1 in this block
            h1s[(size_t)m * NU_ + u] = hh;   // state for next step
        }
    }
    __syncthreads();

    // ---------------- cell 1 ----------------
    a0 = vzero; a1 = vzero; a2 = vzero; a3 = vzero;
    {   // h1' @ Wih1.T (h1' from LDS) -- K = 128
        const _Float16* arow = h1_l + (size_t)col * NU_;
        const _Float16* w0r  = Wih1 + (size_t)(0 * NU_ + u0 + col) * NU_;
        const _Float16* w1r  = Wih1 + (size_t)(1 * NU_ + u0 + col) * NU_;
        const _Float16* w2r  = Wih1 + (size_t)(2 * NU_ + u0 + col) * NU_;
        const _Float16* w3r  = Wih1 + (size_t)(3 * NU_ + u0 + col) * NU_;
        for (int k0 = 0; k0 < NU_; k0 += 32) {
            v16h a = ldfrag(arow, k0, lane);
            a0 = wmma_f16(a, ldfrag(w0r, k0, lane), a0);
            a1 = wmma_f16(a, ldfrag(w1r, k0, lane), a1);
            a2 = wmma_f16(a, ldfrag(w2r, k0, lane), a2);
            a3 = wmma_f16(a, ldfrag(w3r, k0, lane), a3);
        }
    }
    {   // h2 @ Whh1.T -- K = 128
        const _Float16* arow = h2s + (size_t)(m0 + col) * NU_;
        const _Float16* w0r  = Whh1 + (size_t)(0 * NU_ + u0 + col) * NU_;
        const _Float16* w1r  = Whh1 + (size_t)(1 * NU_ + u0 + col) * NU_;
        const _Float16* w2r  = Whh1 + (size_t)(2 * NU_ + u0 + col) * NU_;
        const _Float16* w3r  = Whh1 + (size_t)(3 * NU_ + u0 + col) * NU_;
        for (int k0 = 0; k0 < NU_; k0 += 32) {
            v16h a = ldfrag(arow, k0, lane);
            a0 = wmma_f16(a, ldfrag(w0r, k0, lane), a0);
            a1 = wmma_f16(a, ldfrag(w1r, k0, lane), a1);
            a2 = wmma_f16(a, ldfrag(w2r, k0, lane), a2);
            a3 = wmma_f16(a, ldfrag(w3r, k0, lane), a3);
        }
    }
    __syncthreads();   // all waves done reading h2s
    {
        int u = u0 + col;
        float bi = b1[u], bf = b1[NU_ + u], bg = b1[2 * NU_ + u], bo = b1[3 * NU_ + u];
#pragma unroll
        for (int r = 0; r < 8; ++r) {
            int mr = r + hi;
            int m  = m0 + mr;
            float cprev = c2s[(size_t)m * NU_ + u];
            float c = sigm(a1[r] + bf) * cprev + sigm(a0[r] + bi) * tanhf(a2[r] + bg);
            float h = sigm(a3[r] + bo) * tanhf(c);
            c2s[(size_t)m * NU_ + u] = c;
            _Float16 hh = (_Float16)h;
            h2_l[mr * NU_ + u]       = hh;
            h2s[(size_t)m * NU_ + u] = hh;
        }
    }
    __syncthreads();

    // ---------------- output projection ----------------
    if (tid < 16) {
        int m = m0 + tid;
        float s = outb[0];
        for (int u = 0; u < NU_; ++u)
            s += (float)h2_l[tid * NU_ + u] * outW[u];
        out[(size_t)m * N_ + step] = sigm(s);
    }
}

// ---------------------------------------------------------------------------
extern "C" void kernel_launch(void* const* d_in, const int* in_sizes, int n_in,
                              void* d_out, int out_size, void* d_ws, size_t ws_size,
                              hipStream_t stream) {
    const float* note    = (const float*)d_in[0];
    const float* targets = (const float*)d_in[1];
    const float* t_Wih0  = (const float*)d_in[2];
    // d_in[3] = t_Whh0 : dead (h=0)
    const float* t_b0    = (const float*)d_in[4];
    const float* t_Wih1  = (const float*)d_in[5];
    // d_in[6] = t_Whh1 : dead
    const float* t_b1    = (const float*)d_in[7];
    const float* n_Wih0  = (const float*)d_in[8];
    const float* n_Whh0  = (const float*)d_in[9];
    const float* n_b0    = (const float*)d_in[10];
    const float* n_Wih1  = (const float*)d_in[11];
    const float* n_Whh1  = (const float*)d_in[12];
    const float* n_b1    = (const float*)d_in[13];
    const float* out_W   = (const float*)d_in[14];
    const float* out_b   = (const float*)d_in[15];
    float* out = (float*)d_out;

    char* ws = (char*)d_ws;
    size_t off = 0;
    auto alloc = [&](size_t bytes) -> char* {
        char* p = ws + off;
        off = (off + bytes + 255) & ~(size_t)255;
        return p;
    };
    _Float16* rnn16  = (_Float16*)alloc((size_t)ROWS * K1 * 2);
    _Float16* tW0_16 = (_Float16*)alloc((size_t)4 * TU_ * K1 * 2);
    _Float16* tW1_16 = (_Float16*)alloc((size_t)4 * TU_ * TU_ * 2);
    _Float16* h1t16  = (_Float16*)alloc((size_t)ROWS * TU_ * 2);
    _Float16* feats  = (_Float16*)alloc((size_t)ROWS * TU_ * 2);
    _Float16* nW0_16 = (_Float16*)alloc((size_t)4 * NU_ * TU_ * 2);
    float*    wlast  = (float*)   alloc((size_t)4 * NU_ * 4);
    _Float16* nWh0   = (_Float16*)alloc((size_t)4 * NU_ * NU_ * 2);
    _Float16* nW1_16 = (_Float16*)alloc((size_t)4 * NU_ * NU_ * 2);
    _Float16* nWh1   = (_Float16*)alloc((size_t)4 * NU_ * NU_ * 2);
    _Float16* h1s    = (_Float16*)alloc((size_t)B_ * NU_ * 2);
    _Float16* h2s    = (_Float16*)alloc((size_t)B_ * NU_ * 2);
    float*    c1s    = (float*)   alloc((size_t)B_ * NU_ * 4);
    float*    c2s    = (float*)   alloc((size_t)B_ * NU_ * 4);
    (void)ws_size; (void)in_sizes; (void)n_in; (void)out_size;

    // --- build input features (f16, K padded to 64) ---
    build_rnn_in_kernel<<<(ROWS * K1 + 255) / 256, 256, 0, stream>>>(note, rnn16);

    // --- convert weights to f16 (B stored as W[n][k] row-major) ---
    cvt_pad_kernel<<<(1024 * 64 + 255) / 256, 256, 0, stream>>>(t_Wih0, tW0_16, 1024, 50, 50, 64);
    cvt_pad_kernel<<<(1024 * 256 + 255) / 256, 256, 0, stream>>>(t_Wih1, tW1_16, 1024, 256, 256, 256);
    cvt_pad_kernel<<<(512 * 256 + 255) / 256, 256, 0, stream>>>(n_Wih0, nW0_16, 512, 257, 256, 256);
    extract_col_kernel<<<2, 256, 0, stream>>>(n_Wih0, wlast, 512, 257, 256);
    cvt_pad_kernel<<<(512 * 128 + 255) / 256, 256, 0, stream>>>(n_Whh0, nWh0, 512, 128, 128, 128);
    cvt_pad_kernel<<<(512 * 128 + 255) / 256, 256, 0, stream>>>(n_Wih1, nW1_16, 512, 128, 128, 128);
    cvt_pad_kernel<<<(512 * 128 + 255) / 256, 256, 0, stream>>>(n_Whh1, nWh1, 512, 128, 128, 128);

    // --- time-axis LSTM layers (zero state, f-gate dead): two WMMA GEMMs ---
    const int tl_blocks = (ROWS / 16) * (TU_ / 16) / 8;   // 6144 blocks x 8 waves
    time_lstm_kernel<<<tl_blocks, 256, 0, stream>>>(rnn16, tW0_16, t_b0, h1t16, K1);
    time_lstm_kernel<<<tl_blocks, 256, 0, stream>>>(h1t16, tW1_16, t_b1, feats, TU_);

    // --- note-axis scan: 48 sequential fused two-cell steps ---
    zero_state_kernel<<<(B_ * NU_ + 255) / 256, 256, 0, stream>>>(h1s, h2s, c1s, c2s);
    for (int step = 0; step < N_; ++step) {
        note_step_kernel<<<B_ / 16, 256, 0, stream>>>(
            feats, targets, nW0_16, wlast, nWh0, n_b0, nW1_16, nWh1, n_b1,
            out_W, out_b, h1s, c1s, h2s, c2s, out, step);
    }
}
